// CoAttention_33672543601323
// MI455X (gfx1250) — matrix operand order repaired and checked by reading, not compiled, not run on hardware
//
#include <hip/hip_runtime.h>
#include <hip/hip_bf16.h>
#include <stdint.h>

// Problem constants (from reference setup_inputs)
#define B_  64
#define NS  512
#define NC  512
#define DD  256
#define KK  128

typedef __bf16 bf16;
typedef __attribute__((ext_vector_type(16))) __bf16 v16bf;
typedef __attribute__((ext_vector_type(8)))  __bf16 v8bf;
typedef __attribute__((ext_vector_type(8)))  float   v8f;

union ABfrag { v16bf v; v8bf h[2]; };

#define MAXNORM (1.0f - 1e-5f)
#define MINNRM  (1e-15f)

__device__ __forceinline__ float artanh_c(float x) {
  x = fminf(fmaxf(x, -1.0f + 1e-7f), 1.0f - 1e-7f);
  return 0.5f * (log1pf(x) - log1pf(-x));
}

__device__ __forceinline__ bf16 f2b(float f) {
  union { float f; unsigned u; } x; x.f = f;
  unsigned r = x.u + 0x7FFFu + ((x.u >> 16) & 1u);   // RNE
  unsigned short h = (unsigned short)(r >> 16);
  return __builtin_bit_cast(bf16, h);
}

template<int NT>
__device__ __forceinline__ float blk_sum(float v, float* sm) {
  int t = threadIdx.x;
  sm[t] = v; __syncthreads();
  #pragma unroll
  for (int s = NT >> 1; s > 0; s >>= 1) {
    if (t < s) sm[t] += sm[t + s];
    __syncthreads();
  }
  float r = sm[0]; __syncthreads();
  return r;
}

template<int NT>
__device__ __forceinline__ float blk_max(float v, float* sm) {
  int t = threadIdx.x;
  sm[t] = v; __syncthreads();
  #pragma unroll
  for (int s = NT >> 1; s > 0; s >>= 1) {
    if (t < s) sm[t] = fmaxf(sm[t], sm[t + s]);
    __syncthreads();
  }
  float r = sm[0]; __syncthreads();
  return r;
}

// ---------------------------------------------------------------------------
// Generic batched C = A * B^T   (A:[M,Kd] bf16 rowmajor, B:[N,Kd] bf16 rowmajor)
// Block = 8 waves (2Mx4N), each wave a 32x32 tile (4 accumulators, A/B
// fragments reused 2x each -> 16 FLOP/byte), block tile 64x128.
// K-loop is register double-buffered: next-k fragments are issued before the
// current WMMAs so s_wait_loadcnt only covers in-flight (not consumed) loads.
// EPI==0: store f32 C.  EPI==1: t = tanh(acc), store bf16 to Cbf[m,n] and
// CbfT[n,m] (fused L_e / L_e^T production for GEMM-3).
// Lane layouts (C per CDNA5 ISA; A/B use this code's own k order):
//   A 16x32 bf16: lane m=L%16, K = (L/16)*16 + 0..15 (contiguous, same as B)
//   B 32x16 bf16: lane n=L%16, K = (L/16)*16 + 0..15 (contiguous; not the ISA table order)
//   C 16x16 f32 : lane n=L%16, rows m=(L/16)*8 + r
// ---------------------------------------------------------------------------
template<int EPI>
__global__ __launch_bounds__(256)
void gemm_bt_wmma(const bf16* __restrict__ A, const bf16* __restrict__ B,
                  float* __restrict__ C, bf16* __restrict__ Cbf,
                  bf16* __restrict__ CbfT,
                  int M, int N, int Kd, long sA, long sB, long sC) {
  const int b    = blockIdx.z;
  A += (long)b * sA;
  B += (long)b * sB;
  const int wave = threadIdx.x >> 5;
  const int lane = threadIdx.x & 31;
  const int wm   = wave >> 2;            // 0..1
  const int wn   = wave & 3;             // 0..3
  const int row0 = blockIdx.y * 64 + wm * 32;
  const int col0 = blockIdx.x * 128 + wn * 32;
  const int lh   = lane >> 4;            // lane half
  const int lm   = lane & 15;

  const bf16* pa0 = A + (long)(row0 + lm) * Kd + (lh << 4);
  const bf16* pa1 = pa0 + (long)16 * Kd;
  const bf16* pb0 = B + (long)(col0 + lm) * Kd + (lh << 4);
  const bf16* pb1 = pb0 + (long)16 * Kd;

  v8f acc00 = {}, acc01 = {}, acc10 = {}, acc11 = {};

  // prologue: fragments for k = 0
  ABfrag a0, a1;
  a0.h[0] = *(const v8bf*)(pa0);  a0.h[1] = *(const v8bf*)(pa0 + 8);
  a1.h[0] = *(const v8bf*)(pa1);  a1.h[1] = *(const v8bf*)(pa1 + 8);
  v16bf b0 = *(const v16bf*)(pb0);
  v16bf b1 = *(const v16bf*)(pb1);

  for (int k = 32; k <= Kd; k += 32) {
    ABfrag a0n = a0, a1n = a1;
    v16bf  b0n = b0, b1n = b1;
    if (k < Kd) {                        // uniform condition (no divergence)
      a0n.h[0] = *(const v8bf*)(pa0 + k);  a0n.h[1] = *(const v8bf*)(pa0 + k + 8);
      a1n.h[0] = *(const v8bf*)(pa1 + k);  a1n.h[1] = *(const v8bf*)(pa1 + k + 8);
      b0n = *(const v16bf*)(pb0 + k);
      b1n = *(const v16bf*)(pb1 + k);
      if (k + 96 < Kd) {                 // global_prefetch_b8 (speculative)
        __builtin_prefetch(pa0 + k + 96, 0, 1);
        __builtin_prefetch(pa1 + k + 96, 0, 1);
        __builtin_prefetch(pb0 + k + 96, 0, 1);
        __builtin_prefetch(pb1 + k + 96, 0, 1);
      }
    }
    acc00 = __builtin_amdgcn_wmma_f32_16x16x32_bf16(false, a0.v, false, b0, (short)0, acc00, false, false);
    acc01 = __builtin_amdgcn_wmma_f32_16x16x32_bf16(false, a0.v, false, b1, (short)0, acc01, false, false);
    acc10 = __builtin_amdgcn_wmma_f32_16x16x32_bf16(false, a1.v, false, b0, (short)0, acc10, false, false);
    acc11 = __builtin_amdgcn_wmma_f32_16x16x32_bf16(false, a1.v, false, b1, (short)0, acc11, false, false);
    a0 = a0n; a1 = a1n; b0 = b0n; b1 = b1n;
  }

  if (EPI == 0) {
    float* pc = C + (long)b * sC;
    #pragma unroll
    for (int tm = 0; tm < 2; ++tm) {
      #pragma unroll
      for (int tn = 0; tn < 2; ++tn) {
        const v8f& acc = tm == 0 ? (tn == 0 ? acc00 : acc01)
                                 : (tn == 0 ? acc10 : acc11);
        int cn = col0 + tn * 16 + lm;
        #pragma unroll
        for (int r = 0; r < 8; ++r) {
          int cm = row0 + tm * 16 + (lh << 3) + r;
          pc[(long)cm * N + cn] = acc[r];
        }
      }
    }
  } else {
    const long ob = (long)b * sC;
    #pragma unroll
    for (int tm = 0; tm < 2; ++tm) {
      #pragma unroll
      for (int tn = 0; tn < 2; ++tn) {
        const v8f& acc = tm == 0 ? (tn == 0 ? acc00 : acc01)
                                 : (tn == 0 ? acc10 : acc11);
        int cn = col0 + tn * 16 + lm;
        #pragma unroll
        for (int r = 0; r < 8; ++r) {
          int cm = row0 + tm * 16 + (lh << 3) + r;
          float t = tanhf(acc[r]);
          Cbf[ob + (long)cm * N + cn]  = f2b(t);
          CbfT[ob + (long)cn * M + cm] = f2b(t);
        }
      }
    }
  }
}

// --------------------------------------------------------------------------
// f32 -> bf16 convert (weights)
__global__ void k_f2bf(const float* __restrict__ x, bf16* __restrict__ y, int n) {
  int i = blockIdx.x * blockDim.x + threadIdx.x;
  if (i < n) y[i] = f2b(x[i]);
}

// Per-(b,d) column stats over rows: beta (projx of X^T rows), gamma = logmap0
// factor of projx'd row * beta.  X:[B,rows,D].  grid=B, block=DD.
__global__ __launch_bounds__(DD)
void k_colstats(const float* __restrict__ X, int rows,
                float* __restrict__ beta, float* __restrict__ gamma) {
  int b = blockIdx.x, d = threadIdx.x;
  const float* p = X + (long)b * rows * DD + d;
  float ss = 0.f;
  for (int j = 0; j < rows; ++j) { float v = p[(long)j * DD]; ss += v * v; }
  float n  = fmaxf(sqrtf(ss), MINNRM);
  float be = (n > MAXNORM) ? (MAXNORM / n) : 1.0f;
  float m  = fmaxf(be * n, MINNRM);
  beta[b * DD + d]  = be;
  gamma[b * DD + d] = (artanh_c(m) / m) * be;
}

// Per-row prep: bf16 copy, beta-scaled bf16 copy, row logmap0 factor (lam),
// plain row norm (xnp), scaled row norm (xns).  grid = B*rows, block=DD.
__global__ __launch_bounds__(DD)
void k_prep(const float* __restrict__ X, const float* __restrict__ beta,
            int rowsPerB, bf16* __restrict__ Xbf, bf16* __restrict__ X2bf,
            float* __restrict__ lam, float* __restrict__ xnp,
            float* __restrict__ xns) {
  __shared__ float sm[DD];
  int row = blockIdx.x, t = threadIdx.x, b = row / rowsPerB;
  long idx = (long)row * DD + t;
  float v  = X[idx];
  float v2 = beta[b * DD + t] * v;
  Xbf[idx]  = f2b(v);
  X2bf[idx] = f2b(v2);
  float sp = blk_sum<DD>(v * v, sm);
  float s2 = blk_sum<DD>(v2 * v2, sm);
  if (t == 0) {
    float np = fmaxf(sqrtf(sp), MINNRM);
    xnp[row] = np;
    lam[row] = artanh_c(np) / np;
    xns[row] = fmaxf(sqrtf(s2), MINNRM);
  }
}

// Post GEMM-1: full mobius_matvec scale + projx + logmap0 (all row-scalars),
// fold gamma_s(d), emit bf16 Le1'.  grid = B*NC, block = DD.
__global__ __launch_bounds__(DD)
void k_mobscale_d(const float* __restrict__ mx, const float* __restrict__ xnv,
                  const float* __restrict__ gamma_s, int rowsPerB,
                  bf16* __restrict__ out) {
  __shared__ float sm[DD];
  int row = blockIdx.x, t = threadIdx.x, b = row / rowsPerB;
  long idx = (long)row * DD + t;
  float v  = mx[idx];
  float ss = blk_sum<DD>(v * v, sm);
  float mxn = fmaxf(sqrtf(ss), MINNRM);
  float xn  = xnv[row];
  float t1  = tanhf(mxn / xn * artanh_c(xn));
  float f0  = t1 / mxn;                         // mobius_matvec scale
  float p   = (t1 > MAXNORM) ? (MAXNORM / t1) : 1.0f;  // projx
  float n2  = fmaxf(t1 * p, MINNRM);
  float f   = f0 * p * (artanh_c(n2) / n2);     // logmap0
  out[idx]  = f2b(f * v * gamma_s[b * DD + t]); // fold diag(gamma_s)
}

// Post GEMM-2: mobius_matvec row scale over K, write TRANSPOSED raw result
// [B,K,rowsPerB].  grid = B*rowsPerB, block = KK.
__global__ __launch_bounds__(KK)
void k_mobscale_k(const float* __restrict__ mx, const float* __restrict__ xnv,
                  int rowsPerB, float* __restrict__ outT) {
  __shared__ float sm[KK];
  int row = blockIdx.x, t = threadIdx.x;
  int b = row / rowsPerB, j = row % rowsPerB;
  float v  = mx[(long)row * KK + t];
  float ss = blk_sum<KK>(v * v, sm);
  float mxn = fmaxf(sqrtf(ss), MINNRM);
  float xn  = xnv[row];
  float f   = tanhf(mxn / xn * artanh_c(xn)) / mxn;
  outT[((long)b * KK + t) * rowsPerB + j] = f * v;
}

// A_s / A_c finalize per (b,k) row of length 512: projx -> Aproj (f32, kept
// for mobius_add), logmap0 -> lA (bf16 GEMM operand).  grid=B*K, block=256.
__global__ __launch_bounds__(256)
void k_colfin(const float* __restrict__ raw, float* __restrict__ proj,
              bf16* __restrict__ lA) {
  __shared__ float sm[256];
  int t = threadIdx.x;
  long base = (long)blockIdx.x * 512;
  float v0 = raw[base + t], v1 = raw[base + t + 256];
  float ss = blk_sum<256>(v0 * v0 + v1 * v1, sm);
  float n  = fmaxf(sqrtf(ss), MINNRM);
  float p  = (n > MAXNORM) ? (MAXNORM / n) : 1.0f;
  float n2 = fmaxf(n * p, MINNRM);
  float ls = artanh_c(n2) / n2;
  proj[base + t]       = p * v0;
  proj[base + t + 256] = p * v1;
  lA[base + t]         = f2b(ls * p * v0);
  lA[base + t + 256]   = f2b(ls * p * v1);
}

// H finalize per (b,k) row (len 512):
// y=projx(expmap0(u)); h=projx(mobius_add(x,y)); H=projx(expmap0(tanh(logmap0(h))))
__global__ __launch_bounds__(256)
void k_hfin(const float* __restrict__ xproj, const float* __restrict__ u,
            float* __restrict__ H) {
  __shared__ float sm[256];
  int t = threadIdx.x;
  long base = (long)blockIdx.x * 512;
  float x0 = xproj[base + t], x1 = xproj[base + t + 256];
  float u0 = u[base + t],     u1 = u[base + t + 256];
  float sx  = blk_sum<256>(x0 * x0 + x1 * x1, sm);
  float su  = blk_sum<256>(u0 * u0 + u1 * u1, sm);
  float sxu = blk_sum<256>(x0 * u0 + x1 * u1, sm);
  float nu = fmaxf(sqrtf(su), MINNRM);
  float tu = tanhf(nu);
  float pu = (tu > MAXNORM) ? (MAXNORM / tu) : 1.0f;
  float fy = (tu / nu) * pu;                 // y = fy * u
  float x2 = sx, y2 = (tu * pu) * (tu * pu), xy = sxu * fy;
  float a  = 1.0f + 2.0f * xy + y2;
  float bb = 1.0f - x2;
  float den = fmaxf(1.0f + 2.0f * xy + x2 * y2, MINNRM);
  float h0 = (a * x0 + bb * fy * u0) / den;
  float h1 = (a * x1 + bb * fy * u1) / den;
  float sh = blk_sum<256>(h0 * h0 + h1 * h1, sm);
  float nh = fmaxf(sqrtf(sh), MINNRM);
  float ph = (nh > MAXNORM) ? (MAXNORM / nh) : 1.0f;
  float n1 = fmaxf(nh * ph, MINNRM);
  float l  = artanh_c(n1) / n1;
  float t0 = tanhf(l * ph * h0);
  float t1 = tanhf(l * ph * h1);
  float st = blk_sum<256>(t0 * t0 + t1 * t1, sm);
  float nt = fmaxf(sqrtf(st), MINNRM);
  float tt = tanhf(nt);
  float pt = (tt > MAXNORM) ? (MAXNORM / tt) : 1.0f;
  float fo = (tt / nt) * pt;
  H[base + t]       = fo * t0;
  H[base + t + 256] = fo * t1;
}

// Attention: per b, v_j = mobius_matvec(wh, H[:,j]) over K; then row (len 512):
// projx -> logmap0 -> softmax -> A = projx(expmap0(sm)); smlog = logmap0(A).
__global__ __launch_bounds__(512)
void k_attn(const float* __restrict__ H, const float* __restrict__ wh,
            float* __restrict__ Aout, float* __restrict__ smlog) {
  __shared__ float sm[512];
  int b = blockIdx.x, j = threadIdx.x;
  const float* h = H + (long)b * KK * 512;
  float xx = 0.f, mx = 0.f;
  for (int k = 0; k < KK; ++k) {
    float v = h[(long)k * 512 + j];
    xx += v * v; mx += wh[k] * v;
  }
  float xn  = fmaxf(sqrtf(xx), MINNRM);
  float mxn = fmaxf(fabsf(mx), MINNRM);
  float v   = tanhf(mxn / xn * artanh_c(xn)) * (mx / mxn);
  // projx over Ns
  float sv = blk_sum<512>(v * v, sm);
  float n  = fmaxf(sqrtf(sv), MINNRM);
  float p  = (n > MAXNORM) ? (MAXNORM / n) : 1.0f;
  float n1 = fmaxf(n * p, MINNRM);
  float w  = (artanh_c(n1) / n1) * p * v;     // logmap0(As)
  // softmax
  float wmax = blk_max<512>(w, sm);
  float e    = expf(w - wmax);
  float se   = blk_sum<512>(e, sm);
  float smx  = e / se;
  // A = projx(expmap0(sm))
  float ssm = blk_sum<512>(smx * smx, sm);
  float nsm = fmaxf(sqrtf(ssm), MINNRM);
  float ts  = tanhf(nsm);
  float ps  = (ts > MAXNORM) ? (MAXNORM / ts) : 1.0f;
  float fA  = (ts / nsm) * ps;
  float aj  = fA * smx;
  Aout[b * 512 + j] = aj;
  float nA  = fmaxf(ts * ps, MINNRM);
  smlog[b * 512 + j] = (artanh_c(nA) / nA) * aj;   // logmap0(A) (== smx if ps==1)
}

// co_s / co_c GEMV + concat.  grid=B, block=DD.
__global__ __launch_bounds__(DD)
void k_co(const float* __restrict__ s, const float* __restrict__ c,
          const float* __restrict__ smls, const float* __restrict__ lams,
          const float* __restrict__ smlc, const float* __restrict__ lamc,
          float* __restrict__ out) {
  int b = blockIdx.x, d = threadIdx.x;
  const float* sp = s + (long)b * NS * DD;
  const float* cp = c + (long)b * NC * DD;
  float as = 0.f, ac = 0.f;
  for (int j = 0; j < NS; ++j)
    as += smls[b * NS + j] * lams[b * NS + j] * sp[(long)j * DD + d];
  for (int j = 0; j < NC; ++j)
    ac += smlc[b * NC + j] * lamc[b * NC + j] * cp[(long)j * DD + d];
  out[(long)b * 2 * DD + d]      = as;
  out[(long)b * 2 * DD + DD + d] = ac;
}

// ---------------------------------------------------------------------------
extern "C" void kernel_launch(void* const* d_in, const int* in_sizes, int n_in,
                              void* d_out, int out_size, void* d_ws, size_t ws_size,
                              hipStream_t stream) {
  const float* s   = (const float*)d_in[0];
  const float* c   = (const float*)d_in[1];
  const float* Wl  = (const float*)d_in[2];
  const float* Wc  = (const float*)d_in[3];
  const float* Ws  = (const float*)d_in[4];
  const float* whs = (const float*)d_in[5];
  const float* whc = (const float*)d_in[6];
  float* out = (float*)d_out;

  uintptr_t cur = (uintptr_t)d_ws;
  auto alloc = [&](size_t bytes) -> void* {
    cur = (cur + 255) & ~(uintptr_t)255;
    void* p = (void*)cur; cur += bytes; return p;
  };
  const size_t ND  = (size_t)B_ * NS * DD;   // 8.4M elems
  const size_t NNS = (size_t)B_ * NC * NS;   // 16.8M elems
  const size_t NK  = (size_t)B_ * NS * KK;   // 4.2M elems

  bf16* s_bf  = (bf16*)alloc(ND * 2);
  bf16* c_bf  = (bf16*)alloc(ND * 2);
  bf16* S2_bf = (bf16*)alloc(ND * 2);
  bf16* C2_bf = (bf16*)alloc(ND * 2);
  bf16* Wl_bf = (bf16*)alloc((size_t)DD * DD * 2);
  bf16* Wc_bf = (bf16*)alloc((size_t)KK * DD * 2);
  bf16* Ws_bf = (bf16*)alloc((size_t)KK * DD * 2);
  float* beta_s  = (float*)alloc((size_t)B_ * DD * 4);
  float* gamma_s = (float*)alloc((size_t)B_ * DD * 4);
  float* beta_c  = (float*)alloc((size_t)B_ * DD * 4);
  float* gamma_c = (float*)alloc((size_t)B_ * DD * 4);
  float* lam_s = (float*)alloc((size_t)B_ * NS * 4);
  float* xnp_s = (float*)alloc((size_t)B_ * NS * 4);
  float* xns_s = (float*)alloc((size_t)B_ * NS * 4);
  float* lam_c = (float*)alloc((size_t)B_ * NC * 4);
  float* xnp_c = (float*)alloc((size_t)B_ * NC * 4);
  float* xns_c = (float*)alloc((size_t)B_ * NC * 4);
  float* g1   = (float*)alloc(ND * 4);
  bf16*  Le1  = (bf16*)alloc(ND * 2);
  bf16*  Le   = (bf16*)alloc(NNS * 2);
  bf16*  LeT  = (bf16*)alloc(NNS * 2);
  float* g2s  = (float*)alloc(NK * 4);
  float* g2c  = (float*)alloc(NK * 4);
  float* Asraw = (float*)alloc(NK * 4);
  float* Acraw = (float*)alloc(NK * 4);
  float* Asproj = (float*)alloc(NK * 4);
  float* Acproj = (float*)alloc(NK * 4);
  bf16*  lAs = (bf16*)alloc(NK * 2);
  bf16*  lAc = (bf16*)alloc(NK * 2);
  float* g4 = (float*)alloc(NK * 4);
  float* g5 = (float*)alloc(NK * 4);
  float* Hs = (float*)alloc(NK * 4);
  float* Hc = (float*)alloc(NK * 4);
  float* smls = (float*)alloc((size_t)B_ * NS * 4);
  float* smlc = (float*)alloc((size_t)B_ * NC * 4);
  (void)ws_size; (void)in_sizes; (void)n_in; (void)out_size;

  // 1. weights -> bf16
  k_f2bf<<<(DD * DD + 255) / 256, 256, 0, stream>>>(Wl, Wl_bf, DD * DD);
  k_f2bf<<<(KK * DD + 255) / 256, 256, 0, stream>>>(Wc, Wc_bf, KK * DD);
  k_f2bf<<<(KK * DD + 255) / 256, 256, 0, stream>>>(Ws, Ws_bf, KK * DD);

  // 2. column stats (projx of transposed reps + logmap0 factor)
  k_colstats<<<B_, DD, 0, stream>>>(s, NS, beta_s, gamma_s);
  k_colstats<<<B_, DD, 0, stream>>>(c, NC, beta_c, gamma_c);

  // 3. row prep: bf16 copies, scaled copies, norms
  k_prep<<<B_ * NS, DD, 0, stream>>>(s, beta_s, NS, s_bf, S2_bf, lam_s, xnp_s, xns_s);
  k_prep<<<B_ * NC, DD, 0, stream>>>(c, beta_c, NC, c_bf, C2_bf, lam_c, xnp_c, xns_c);

  // 4. GEMM1: c @ Wl^T   [B*Nc, D] x [D, D]   (block tile 64x128)
  gemm_bt_wmma<0><<<dim3(DD / 128, (B_ * NC) / 64, 1), 256, 0, stream>>>(
      c_bf, Wl_bf, g1, nullptr, nullptr, B_ * NC, DD, DD, 0, 0, 0);

  // 5. Le1' = logmap0(projx(mobius(g1))) * diag(gamma_s)  (bf16)
  k_mobscale_d<<<B_ * NC, DD, 0, stream>>>(g1, xnp_c, gamma_s, NC, Le1);

  // 6. GEMM3 batched: M = Le1' @ s^T, fused tanh -> L_e and L_e^T (bf16)
  gemm_bt_wmma<1><<<dim3(NS / 128, NC / 64, B_), 256, 0, stream>>>(
      Le1, s_bf, nullptr, Le, LeT, NC, NS, DD,
      (long)NC * DD, (long)NS * DD, (long)NC * NS);

  // 7. GEMM2a/2b: S2 @ Ws^T, C2 @ Wc^T
  gemm_bt_wmma<0><<<dim3(KK / 128, (B_ * NS) / 64, 1), 256, 0, stream>>>(
      S2_bf, Ws_bf, g2s, nullptr, nullptr, B_ * NS, KK, DD, 0, 0, 0);
  gemm_bt_wmma<0><<<dim3(KK / 128, (B_ * NC) / 64, 1), 256, 0, stream>>>(
      C2_bf, Wc_bf, g2c, nullptr, nullptr, B_ * NC, KK, DD, 0, 0, 0);

  // 8. mobius row-scale + transpose -> raw A_s, A_c  [B,K,Ns]
  k_mobscale_k<<<B_ * NS, KK, 0, stream>>>(g2s, xns_s, NS, Asraw);
  k_mobscale_k<<<B_ * NC, KK, 0, stream>>>(g2c, xns_c, NC, Acraw);

  // 9. projx + logmap0 per (b,k) row
  k_colfin<<<B_ * KK, 256, 0, stream>>>(Asraw, Asproj, lAs);
  k_colfin<<<B_ * KK, 256, 0, stream>>>(Acraw, Acproj, lAc);

  // 10. GEMM4: lA_c @ L_e  (via LeT as B operand)  -> g4 [B,K,Ns]
  gemm_bt_wmma<0><<<dim3(NS / 128, KK / 64, B_), 256, 0, stream>>>(
      lAc, LeT, g4, nullptr, nullptr, KK, NS, NC,
      (long)KK * NC, (long)NS * NC, (long)KK * NS);
  // 11. GEMM5: lA_s @ L_e^T (via Le as B operand) -> g5 [B,K,Nc]
  gemm_bt_wmma<0><<<dim3(NC / 128, KK / 64, B_), 256, 0, stream>>>(
      lAs, Le, g5, nullptr, nullptr, KK, NC, NS,
      (long)KK * NS, (long)NC * NS, (long)KK * NC);

  // 12. Hs, Hc
  k_hfin<<<B_ * KK, 256, 0, stream>>>(Asproj, g4, Hs);
  k_hfin<<<B_ * KK, 256, 0, stream>>>(Acproj, g5, Hc);

  // 13. attention rows -> As (out), Ac (out), smlog for GEMV
  float* As_out = out + (size_t)B_ * 2 * DD;
  float* Ac_out = As_out + (size_t)B_ * NS;
  k_attn<<<B_, 512, 0, stream>>>(Hs, whs, As_out, smls);
  k_attn<<<B_, 512, 0, stream>>>(Hc, whc, Ac_out, smlc);

  // 14. co_sc = [logmap0(As)@logmap0(s) , logmap0(Ac)@logmap0(c)]
  k_co<<<B_, DD, 0, stream>>>(s, c, smls, lam_s, smlc, lam_c, out);
}
